// NAFBlock_68728066671159
// MI455X (gfx1250) — matrix operand order repaired and checked by reading, not compiled, not run on hardware
//
#include <hip/hip_runtime.h>
#include <cstdint>

typedef __bf16 bf16;
typedef __attribute__((ext_vector_type(16))) __bf16 bf16x16;
typedef __attribute__((ext_vector_type(8)))  float  floatx8;

#define B_  8
#define C_  64
#define HID 128
#define C2  256
#define H_  256
#define W_  256

static __device__ __forceinline__ floatx8 wmma_bf16(bf16x16 a, bf16x16 b, floatx8 c) {
  // D = A(16x32 bf16) x B(32x16 bf16) + C(16x16 f32)
  return __builtin_amdgcn_wmma_f32_16x16x32_bf16(false, a, false, b, (short)0, c, false, false);
}

// ---------------- k0: weight bf16 conversion + pool zero ----------------
__global__ void k0_prep(const float* __restrict__ w1, const float* __restrict__ w3,
                        bf16* __restrict__ w1b, bf16* __restrict__ w3b,
                        float* __restrict__ pool) {
  int i = blockIdx.x * blockDim.x + threadIdx.x;
  if (i < C2 * C_)  w1b[i] = (bf16)w1[i];   // w1: (256,64)
  if (i < C_ * HID) w3b[i] = (bf16)w3[i];   // w3: (64,128)
  if (i < B_ * HID) pool[i] = 0.f;
}

// ---------------- k1: LayerNorm + 1x1 conv (64->256, WMMA) + SimpleGate ----------------
__global__ __launch_bounds__(128) void k1_ln_conv1_gate(
    const float* __restrict__ x, const float* __restrict__ ln_w, const float* __restrict__ ln_b,
    const float* __restrict__ b1, const bf16* __restrict__ w1b, bf16* __restrict__ t1) {
  __shared__ float xs[64][C_ + 1];   // 64 pixels x 64 ch, padded (bank-conflict free)
  __shared__ bf16  xa[64][C_];       // normalized, bf16
  __shared__ bf16  wb[C2][C_];       // w1 bf16 [o][c]
  const int t   = threadIdx.x;
  const int gid = blockIdx.x;
  const int x0  = (gid & 3) * 64;
  const int y   = (gid >> 2) & 255;
  const int b   = gid >> 10;

  { // w1 bf16 -> LDS (8192 dwords)
    const uint32_t* src = (const uint32_t*)w1b;
    uint32_t* dst = (uint32_t*)&wb[0][0];
    #pragma unroll
    for (int k = 0; k < 64; ++k) dst[t + 128 * k] = src[t + 128 * k];
  }
  // x tile (NCHW) -> LDS, coalesced along W
  for (int k = 0; k < 32; ++k) {
    int e = t + 128 * k;
    int c = e >> 6, i = e & 63;
    xs[i][c] = x[(((size_t)b * C_ + c) * H_ + y) * W_ + x0 + i];
  }
  __syncthreads();
  if (t < 64) { // per-pixel LayerNorm over 64 channels
    float mu = 0.f;
    #pragma unroll 8
    for (int c = 0; c < C_; ++c) mu += xs[t][c];
    mu *= (1.f / C_);
    float var = 0.f;
    #pragma unroll 8
    for (int c = 0; c < C_; ++c) { float d = xs[t][c] - mu; var += d * d; }
    var *= (1.f / C_);
    float r = rsqrtf(var + 1e-6f);
    for (int c = 0; c < C_; ++c)
      xa[t][c] = (bf16)(((xs[t][c] - mu) * r) * ln_w[c] + ln_b[c]);
  }
  __syncthreads();

  const int wv = t >> 5, l = t & 31;
  const int hi = l >> 4;           // lane half
  const int n  = l & 15;           // N index / A row
  const int mrow = 16 * wv + n;    // pixel row of A
  // A fragments: K 0..31 and 32..63 (contiguous k = 16*hi + j, matching B)
  bf16x16 a0, a1;
  #pragma unroll
  for (int j = 0; j < 16; ++j) {
    int kk = 16 * hi + j;
    a0[j] = xa[mrow][kk];
    a1[j] = xa[mrow][32 + kk];
  }
  const size_t pixbase = (size_t)(b * H_ + y) * W_ + x0;
  for (int jn = 0; jn < 8; ++jn) { // gate pairs: N-tile jn (ch) with jn+8 (ch+128)
    floatx8 acc0 = {0.f,0.f,0.f,0.f,0.f,0.f,0.f,0.f};
    floatx8 acc1 = {0.f,0.f,0.f,0.f,0.f,0.f,0.f,0.f};
    bf16x16 bf0, bf1;
    #pragma unroll
    for (int j = 0; j < 16; ++j) {           // K-half 0: B rows 0..31
      bf0[j] = wb[16 * jn + n][16 * hi + j];
      bf1[j] = wb[16 * (jn + 8) + n][16 * hi + j];
    }
    acc0 = wmma_bf16(a0, bf0, acc0);
    acc1 = wmma_bf16(a0, bf1, acc1);
    #pragma unroll
    for (int j = 0; j < 16; ++j) {           // K-half 1: B rows 32..63
      bf0[j] = wb[16 * jn + n][32 + 16 * hi + j];
      bf1[j] = wb[16 * (jn + 8) + n][32 + 16 * hi + j];
    }
    acc0 = wmma_bf16(a1, bf0, acc0);
    acc1 = wmma_bf16(a1, bf1, acc1);
    int ch = 16 * jn + n;
    float bb0 = b1[ch], bb1 = b1[ch + 128];
    #pragma unroll
    for (int r = 0; r < 8; ++r) {            // C layout: M = r + 8*hi
      int px = 16 * wv + r + 8 * hi;
      float g = (acc0[r] + bb0) * (acc1[r] + bb1);   // SimpleGate
      t1[(pixbase + px) * HID + ch] = (bf16)g;       // NHWC bf16
    }
  }
}

// ---------------- k2: grouped 3x3 conv (2 out/group) + gate + pool partials ----------------
__global__ __launch_bounds__(256) void k2_dwconv_gate_pool(
    const bf16* __restrict__ t1, const float* __restrict__ w2, const float* __restrict__ b2,
    bf16* __restrict__ t2, float* __restrict__ pool) {
  __shared__ float ws2[C2 * 9];
  __shared__ float pacc[HID];
  const int t   = threadIdx.x;
  const int gid = blockIdx.x;
  const int x0  = (gid & 15) * 16;
  const int y   = (gid >> 4) & 255;
  const int b   = gid >> 12;
  #pragma unroll
  for (int j = 0; j < 9; ++j) {
    int e = t + 256 * j;
    if (e < C2 * 9) ws2[e] = w2[e];
  }
  if (t < HID) pacc[t] = 0.f;
  __syncthreads();
  const int c  = t & 127;
  const int pp = t >> 7;
  float wA[9], wB[9];
  #pragma unroll
  for (int j = 0; j < 9; ++j) { wA[j] = ws2[c * 9 + j]; wB[j] = ws2[(c + 128) * 9 + j]; }
  const float bb0 = b2[c], bb1 = b2[c + 128];
  const int ci0 = c >> 1, ci1 = 64 + (c >> 1); // group->input channel mapping
  float lsum = 0.f;
  for (int k = 0; k < 8; ++k) {
    int px = x0 + pp * 8 + k;
    float z0 = bb0, z1 = bb1;
    #pragma unroll
    for (int dy = -1; dy <= 1; ++dy) {
      int yy = y + dy;
      #pragma unroll
      for (int dx = -1; dx <= 1; ++dx) {
        int xx = px + dx;
        int j = (dy + 1) * 3 + (dx + 1);
        if (yy >= 0 && yy < H_ && xx >= 0 && xx < W_) {
          size_t base = ((size_t)(b * H_ + yy) * W_ + xx) * HID;
          z0 += wA[j] * (float)t1[base + ci0];
          z1 += wB[j] * (float)t1[base + ci1];
        }
      }
    }
    float h = z0 * z1;                                   // SimpleGate
    t2[((size_t)(b * H_ + y) * W_ + px) * HID + c] = (bf16)h;
    lsum += h;
  }
  atomicAdd(&pacc[c], lsum);
  __syncthreads();
  if (t < HID) atomicAdd(&pool[b * HID + t], pacc[t]);
}

// ---------------- k3: channel attention (128x128 matvec per batch) ----------------
__global__ __launch_bounds__(128) void k3_attn(
    const float* __restrict__ w_sca, const float* __restrict__ b_sca,
    const float* __restrict__ pool, float* __restrict__ attn) {
  __shared__ float ps[HID];
  const int b = blockIdx.x, o = threadIdx.x;
  ps[o] = pool[b * HID + o] * (1.f / (H_ * W_));
  __syncthreads();
  float s = b_sca[o];
  for (int cc = 0; cc < HID; ++cc) s += w_sca[o * HID + cc] * ps[cc];
  attn[b * HID + o] = s;
}

// ---------------- k4: scale by attn + 1x1 conv (128->64, WMMA) + residual ----------------
__global__ __launch_bounds__(128) void k4_conv3_residual(
    const bf16* __restrict__ t2, const bf16* __restrict__ w3b,
    const float* __restrict__ b3, const float* __restrict__ beta,
    const float* __restrict__ attn, const float* __restrict__ x,
    float* __restrict__ out) {
  __shared__ bf16  ha[64][HID];   // attn-scaled activations, bf16
  __shared__ bf16  wb[C_][HID];   // w3 bf16 [o][c]
  __shared__ float as[HID];
  const int t   = threadIdx.x;
  const int gid = blockIdx.x;
  const int x0  = (gid & 3) * 64;
  const int y   = (gid >> 2) & 255;
  const int b   = gid >> 10;
  {
    const uint32_t* src = (const uint32_t*)w3b;
    uint32_t* dst = (uint32_t*)&wb[0][0];
    #pragma unroll
    for (int k = 0; k < 32; ++k) dst[t + 128 * k] = src[t + 128 * k];
  }
  if (t < HID) as[t] = attn[b * HID + t];
  __syncthreads();
  const size_t pixbase = (size_t)(b * H_ + y) * W_ + x0;
  for (int k = 0; k < 64; ++k) {
    int e = t + 128 * k;
    int px = e >> 7, ch = e & 127;
    ha[px][ch] = (bf16)((float)t2[(pixbase + px) * HID + ch] * as[ch]);
  }
  __syncthreads();
  const int wv = t >> 5, l = t & 31;
  const int hi = l >> 4, n = l & 15;
  const int mrow = 16 * wv + n;
  bf16x16 afr[4];                       // K = 0..127 in 4 halves of 32
  #pragma unroll
  for (int kh = 0; kh < 4; ++kh) {
    #pragma unroll
    for (int j = 0; j < 16; ++j) {
      int kk = 16 * hi + j;
      afr[kh][j] = ha[mrow][32 * kh + kk];
    }
  }
  for (int jn = 0; jn < 4; ++jn) {      // N = 64 out channels
    floatx8 acc = {0.f,0.f,0.f,0.f,0.f,0.f,0.f,0.f};
    #pragma unroll
    for (int kh = 0; kh < 4; ++kh) {
      bf16x16 bfr;
      #pragma unroll
      for (int j = 0; j < 16; ++j) bfr[j] = wb[16 * jn + n][32 * kh + 16 * hi + j];
      acc = wmma_bf16(afr[kh], bfr, acc);
    }
    int ch = 16 * jn + n;
    float bb = b3[ch], bt = beta[ch];
    #pragma unroll
    for (int r = 0; r < 8; ++r) {
      int px = 16 * wv + r + 8 * hi;
      size_t gi = (((size_t)b * C_ + ch) * H_ + y) * W_ + x0 + px;
      out[gi] = x[gi] + (acc[r] + bb) * bt;   // identity + h*beta
    }
  }
}

extern "C" void kernel_launch(void* const* d_in, const int* in_sizes, int n_in,
                              void* d_out, int out_size, void* d_ws, size_t ws_size,
                              hipStream_t stream) {
  const float* x     = (const float*)d_in[0];
  const float* ln_w  = (const float*)d_in[1];
  const float* ln_b  = (const float*)d_in[2];
  const float* w1    = (const float*)d_in[3];
  const float* b1    = (const float*)d_in[4];
  const float* w2    = (const float*)d_in[5];
  const float* b2    = (const float*)d_in[6];
  const float* w_sca = (const float*)d_in[7];
  const float* b_sca = (const float*)d_in[8];
  const float* w3    = (const float*)d_in[9];
  const float* b3    = (const float*)d_in[10];
  const float* beta  = (const float*)d_in[11];
  float* out = (float*)d_out;

  char* wsp = (char*)d_ws;
  const size_t t1_bytes = (size_t)B_ * HID * H_ * W_ * sizeof(bf16); // 128 MB
  bf16*  t1   = (bf16*)wsp;
  bf16*  t2   = (bf16*)(wsp + t1_bytes);
  float* pool = (float*)(wsp + 2 * t1_bytes);
  float* attn = (float*)(wsp + 2 * t1_bytes + 4096);
  bf16*  w1b  = (bf16*)(wsp + 2 * t1_bytes + 8192);
  bf16*  w3b  = (bf16*)(wsp + 2 * t1_bytes + 8192 + 32768);

  k0_prep<<<dim3(128), dim3(256), 0, stream>>>(w1, w3, w1b, w3b, pool);
  k1_ln_conv1_gate<<<dim3(B_ * H_ * (W_ / 64)), dim3(128), 0, stream>>>(x, ln_w, ln_b, b1, w1b, t1);
  k2_dwconv_gate_pool<<<dim3(B_ * H_ * (W_ / 16)), dim3(256), 0, stream>>>(t1, w2, b2, t2, pool);
  k3_attn<<<dim3(B_), dim3(128), 0, stream>>>(w_sca, b_sca, pool, attn);
  k4_conv3_residual<<<dim3(B_ * H_ * (W_ / 64)), dim3(128), 0, stream>>>(t2, w3b, b3, beta, attn, x, out);
}
